// Attention_31954556682759
// MI455X (gfx1250) — compile-verified, matrix operand order checked by reading, not run on hardware
//
#include <hip/hip_runtime.h>
#include <hip/hip_bf16.h>
#include <math.h>

// Problem constants (B=1)
#define S_LEN   4096
#define HID     2048
#define NHEADS  16
#define HDIM    128
#define ROPE    32
#define WIN     1024
#define QKV_N   (3 * HID)

typedef __attribute__((ext_vector_type(16))) __bf16 v16bf;
typedef __attribute__((ext_vector_type(8)))  float  v8f;

union BF16Frag { v16bf v; uint4 q[2]; };
union AccFrag  { v8f v; float f[8]; };

__device__ __forceinline__ unsigned short f2bf(float f) {
    unsigned int u = __float_as_uint(f);
    u += 0x7FFFu + ((u >> 16) & 1u);          // round-to-nearest-even
    return (unsigned short)(u >> 16);
}
__device__ __forceinline__ float bf2f(unsigned short h) {
    return __uint_as_float(((unsigned int)h) << 16);
}

// Load one 16x32 bf16 WMMA A/B fragment from a row-major, K-contiguous ("NT")
// source. Per-lane layout (ISA 7.12.2, 16-bit A 16x32): lane<16 -> row=lane,
// K 0-7 (VGPR0-3) and 16-23 (VGPR4-7); lane>=16 -> row=lane-16, K 8-15 / 24-31.
// Two global_load_b128 per fragment.
__device__ __forceinline__ v16bf load_frag_nt(const unsigned short* base,
                                              long ld, int row0, int k0) {
    const int lane = threadIdx.x & 31;
    const int r    = row0 + (lane & 15);
    const int kh   = (lane >> 4) * 8;
    const unsigned short* p = base + (long)r * ld + k0 + kh;
    BF16Frag f;
    f.q[0] = *(const uint4*)(p);
    f.q[1] = *(const uint4*)(p + 16);
    return f.v;
}

// Same pattern but from LDS (ds_load_b128). ld in elements, k0 = 0.
__device__ __forceinline__ v16bf load_frag_lds(const unsigned short* base, int ld) {
    const int lane = threadIdx.x & 31;
    const int r    = lane & 15;
    const int kh   = (lane >> 4) * 8;
    const unsigned short* p = base + r * ld + kh;
    BF16Frag f;
    f.q[0] = *(const uint4*)(p);
    f.q[1] = *(const uint4*)(p + 16);
    return f.v;
}

__device__ __forceinline__ v8f wmma_bf16(v16bf a, v16bf b, v8f c) {
    return __builtin_amdgcn_wmma_f32_16x16x32_bf16(false, a, false, b,
                                                   (short)0, c, false, false);
}

// ---------------------------------------------------------------- RMSNorm
__global__ __launch_bounds__(256) void rmsnorm_kernel(const float* __restrict__ x,
                                                      const float* __restrict__ w,
                                                      unsigned short* __restrict__ xn) {
    __shared__ float red[256];
    const int row = blockIdx.x;
    const int tid = threadIdx.x;
    const float* xr = x + (long)row * HID;
    float xs[8];
    float ss = 0.f;
#pragma unroll
    for (int i = 0; i < 8; ++i) {
        float v = xr[tid + i * 256];
        xs[i] = v;
        ss += v * v;
    }
    red[tid] = ss;
    __syncthreads();
    for (int st = 128; st > 0; st >>= 1) {
        if (tid < st) red[tid] += red[tid + st];
        __syncthreads();
    }
    const float inv = rsqrtf(red[0] * (1.0f / HID) + 1e-5f);
    unsigned short* o = xn + (long)row * HID;
#pragma unroll
    for (int i = 0; i < 8; ++i)
        o[tid + i * 256] = f2bf(xs[i] * inv * w[tid + i * 256]);
}

// ---------------------------------------------------- f32 -> bf16 weights
__global__ void cvt_bf16_kernel(const float* __restrict__ in,
                                unsigned short* __restrict__ out, long n) {
    long i = (long)blockIdx.x * blockDim.x + threadIdx.x;
    if (i < n) out[i] = f2bf(in[i]);
}

// ------------------------------------------------------------ NT bf16 GEMM
// C[M,N] = A[M,K] * B[N,K]^T ; A,B bf16 row-major (K contiguous), C fp32/bf16.
// 4 waves/block, each wave computes a 64x64 tile (4x4 WMMA accumulators).
template <bool OUT_BF16>
__global__ __launch_bounds__(128) void gemm_nt_kernel(const unsigned short* __restrict__ A,
                                                      const unsigned short* __restrict__ B,
                                                      void* __restrict__ Cp,
                                                      int M, int N, int K) {
    const int lane = threadIdx.x & 31;
    const int wave = threadIdx.x >> 5;
    const int wm = wave >> 1, wn = wave & 1;
    const int m0 = blockIdx.y * 128 + wm * 64;
    const int n0 = blockIdx.x * 128 + wn * 64;

    AccFrag acc[4][4];
#pragma unroll
    for (int i = 0; i < 4; ++i)
#pragma unroll
        for (int j = 0; j < 4; ++j)
#pragma unroll
            for (int r = 0; r < 8; ++r) acc[i][j].f[r] = 0.f;

    for (int k0 = 0; k0 < K; k0 += 32) {
        v16bf af[4], bfm[4];
#pragma unroll
        for (int i = 0; i < 4; ++i) af[i]  = load_frag_nt(A, K, m0 + 16 * i, k0);
#pragma unroll
        for (int j = 0; j < 4; ++j) bfm[j] = load_frag_nt(B, K, n0 + 16 * j, k0);
#pragma unroll
        for (int i = 0; i < 4; ++i)
#pragma unroll
            for (int j = 0; j < 4; ++j)
                acc[i][j].v = wmma_bf16(af[i], bfm[j], acc[i][j].v);
    }

    // C/D layout: VGPR r, lanes 0-15 -> M=r, lanes 16-31 -> M=r+8, N=lane&15.
    const int half = lane >> 4;
    const int cc   = lane & 15;
#pragma unroll
    for (int i = 0; i < 4; ++i)
#pragma unroll
        for (int j = 0; j < 4; ++j)
#pragma unroll
            for (int r = 0; r < 8; ++r) {
                long row = m0 + 16 * i + r + 8 * half;
                long col = n0 + 16 * j + cc;
                if (OUT_BF16)
                    ((unsigned short*)Cp)[row * N + col] = f2bf(acc[i][j].f[r]);
                else
                    ((float*)Cp)[row * N + col] = acc[i][j].f[r];
            }
}

// --------------------------------------------- RoPE + head split/transpose
// qkv bf16 [S, 3H] -> Q,K bf16 [NH,S,HD] (RoPE on d<32), Vt bf16 [NH,HD,S]
__global__ __launch_bounds__(128) void rope_split_kernel(const unsigned short* __restrict__ qkv,
                                                         const float* __restrict__ cosb,
                                                         const float* __restrict__ sinb,
                                                         unsigned short* __restrict__ Q,
                                                         unsigned short* __restrict__ K,
                                                         unsigned short* __restrict__ Vt) {
    const int s = blockIdx.x, h = blockIdx.y, d = threadIdx.x;
    const long base = (long)s * QKV_N + h * HDIM + d;
    float q = bf2f(qkv[base]);
    float k = bf2f(qkv[base + HID]);
    float v = bf2f(qkv[base + 2 * HID]);
    if (d < ROPE) {
        const float c  = cosb[s * ROPE + d];
        const float sn = sinb[s * ROPE + d];
        const int pd   = (d < 16) ? d + 16 : d - 16;
        const long pb  = (long)s * QKV_N + h * HDIM + pd;
        const float qp = bf2f(qkv[pb]);
        const float kp = bf2f(qkv[pb + HID]);
        if (d < 16) { q = q * c - qp * sn; k = k * c - kp * sn; }
        else        { q = q * c + qp * sn; k = k * c + kp * sn; }
    }
    const long hs = (long)h * S_LEN * HDIM + (long)s * HDIM + d;
    Q[hs] = f2bf(q);
    K[hs] = f2bf(k);
    Vt[(long)h * HDIM * S_LEN + (long)d * S_LEN + s] = f2bf(v);
}

// ------------------------------------------------- sliding-window attention
// 4 waves/block; each wave: 16 queries of one head, flash-style over 32-key
// chunks. Scores: 8 WMMA, PV: 8 WMMA (via Vt, NT layout). P staged in LDS.
__global__ __launch_bounds__(128) void attn_kernel(const unsigned short* __restrict__ Q,
                                                   const unsigned short* __restrict__ Kb,
                                                   const unsigned short* __restrict__ Vt,
                                                   unsigned short* __restrict__ Obf) {
    __shared__ unsigned short Plds[4][16 * 32];
    const int lane = threadIdx.x & 31;
    const int wave = threadIdx.x >> 5;
    const int head = blockIdx.y;
    const int q0   = blockIdx.x * 64 + wave * 16;
    const int half = lane >> 4;
    const int nid  = lane & 15;

    const unsigned short* Qh = Q  + (long)head * S_LEN * HDIM;
    const unsigned short* Kh = Kb + (long)head * S_LEN * HDIM;
    const unsigned short* Vh = Vt + (long)head * HDIM * S_LEN;

    v16bf qf[4];
#pragma unroll
    for (int t = 0; t < 4; ++t) qf[t] = load_frag_nt(Qh, HDIM, q0, t * 32);

    AccFrag o[8];
    float mrow[8], lrow[8];
#pragma unroll
    for (int r = 0; r < 8; ++r) { mrow[r] = -3.0e38f; lrow[r] = 0.f; }
#pragma unroll
    for (int dt = 0; dt < 8; ++dt)
#pragma unroll
        for (int r = 0; r < 8; ++r) o[dt].f[r] = 0.f;

    int lo = q0 - WIN; if (lo < 0) lo = 0; lo &= ~31;
    const int hi = q0 + 15;
    const float scale = 0.08838834764831845f; // 1/sqrt(128)

    for (int kc = lo; kc <= hi; kc += 32) {
        AccFrag s0, s1;
#pragma unroll
        for (int r = 0; r < 8; ++r) { s0.f[r] = 0.f; s1.f[r] = 0.f; }
#pragma unroll
        for (int t = 0; t < 4; ++t) {
            v16bf kf = load_frag_nt(Kh, HDIM, kc, t * 32);
            s0.v = wmma_bf16(qf[t], kf, s0.v);
        }
#pragma unroll
        for (int t = 0; t < 4; ++t) {
            v16bf kf = load_frag_nt(Kh, HDIM, kc + 16, t * 32);
            s1.v = wmma_bf16(qf[t], kf, s1.v);
        }

        float p0[8], p1[8];
#pragma unroll
        for (int r = 0; r < 8; ++r) {
            const int i  = q0 + r + 8 * half;
            const int j0 = kc + nid;
            const int j1 = kc + 16 + nid;
            float x0 = (j0 <= i && j0 >= i - WIN) ? s0.f[r] * scale : -3.0e38f;
            float x1 = (j1 <= i && j1 >= i - WIN) ? s1.f[r] * scale : -3.0e38f;
            float t = fmaxf(x0, x1);
            t = fmaxf(t, __shfl_xor(t, 1, 32));
            t = fmaxf(t, __shfl_xor(t, 2, 32));
            t = fmaxf(t, __shfl_xor(t, 4, 32));
            t = fmaxf(t, __shfl_xor(t, 8, 32));
            const float nm   = fmaxf(mrow[r], t);
            const float corr = __expf(mrow[r] - nm);
            mrow[r] = nm;
            const float e0 = (x0 > -1.0e37f) ? __expf(x0 - nm) : 0.f;
            const float e1 = (x1 > -1.0e37f) ? __expf(x1 - nm) : 0.f;
            float rs = e0 + e1;
            rs += __shfl_xor(rs, 1, 32);
            rs += __shfl_xor(rs, 2, 32);
            rs += __shfl_xor(rs, 4, 32);
            rs += __shfl_xor(rs, 8, 32);
            lrow[r] = lrow[r] * corr + rs;
#pragma unroll
            for (int dt = 0; dt < 8; ++dt) o[dt].f[r] *= corr;
            p0[r] = e0; p1[r] = e1;
        }

        // stage P (16x32 bf16) in this wave's private LDS slab
        unsigned short* pl = Plds[wave];
#pragma unroll
        for (int r = 0; r < 8; ++r) {
            const int m = r + 8 * half;
            pl[m * 32 + nid]      = f2bf(p0[r]);
            pl[m * 32 + 16 + nid] = f2bf(p1[r]);
        }
        asm volatile("s_wait_dscnt 0" ::: "memory");
        __builtin_amdgcn_wave_barrier();
        v16bf pf = load_frag_lds(pl, 32);

#pragma unroll
        for (int dt = 0; dt < 8; ++dt) {
            v16bf vf = load_frag_nt(Vh, S_LEN, dt * 16, kc);
            o[dt].v = wmma_bf16(pf, vf, o[dt].v);
        }
    }

#pragma unroll
    for (int r = 0; r < 8; ++r) {
        const float inv = 1.0f / lrow[r];
        const long row = q0 + r + 8 * half;
#pragma unroll
        for (int dt = 0; dt < 8; ++dt)
            Obf[row * HID + head * HDIM + dt * 16 + nid] = f2bf(o[dt].f[r] * inv);
    }
}

// ---------------------------------------------------------------- launcher
extern "C" void kernel_launch(void* const* d_in, const int* in_sizes, int n_in,
                              void* d_out, int out_size, void* d_ws, size_t ws_size,
                              hipStream_t stream) {
    const float* x    = (const float*)d_in[0];   // [1,4096,2048]
    const float* cosb = (const float*)d_in[1];   // [4096,32]
    const float* sinb = (const float*)d_in[2];   // [4096,32]
    const float* nw   = (const float*)d_in[3];   // [2048]
    const float* qkvw = (const float*)d_in[4];   // [6144,2048]
    const float* ow   = (const float*)d_in[5];   // [2048,2048]
    float* out = (float*)d_out;                  // [1,4096,2048]

    char* ws = (char*)d_ws;
    unsigned short* xn      = (unsigned short*)(ws);                 // 16 MB
    unsigned short* qkvw_bf = (unsigned short*)(ws + 16777216L);     // 24 MB
    unsigned short* ow_bf   = (unsigned short*)(ws + 41943040L);     //  8 MB
    unsigned short* qkv_bf  = (unsigned short*)(ws + 50331648L);     // 48 MB
    unsigned short* Qb      = (unsigned short*)(ws + 100663296L);    // 16 MB
    unsigned short* Kb      = (unsigned short*)(ws + 117440512L);    // 16 MB
    unsigned short* Vtb     = (unsigned short*)(ws + 134217728L);    // 16 MB
    unsigned short* attn_o  = (unsigned short*)(ws + 150994944L);    // 16 MB
    (void)in_sizes; (void)n_in; (void)out_size; (void)ws_size;

    // 1) RMSNorm -> bf16
    rmsnorm_kernel<<<S_LEN, 256, 0, stream>>>(x, nw, xn);

    // 2) weights -> bf16
    {
        long n1 = (long)QKV_N * HID;
        cvt_bf16_kernel<<<(unsigned)((n1 + 255) / 256), 256, 0, stream>>>(qkvw, qkvw_bf, n1);
        long n2 = (long)HID * HID;
        cvt_bf16_kernel<<<(unsigned)((n2 + 255) / 256), 256, 0, stream>>>(ow, ow_bf, n2);
    }

    // 3) QKV projection: [4096,2048] x [6144,2048]^T -> bf16 [4096,6144]
    gemm_nt_kernel<true><<<dim3(QKV_N / 128, S_LEN / 128), 128, 0, stream>>>(
        xn, qkvw_bf, qkv_bf, S_LEN, QKV_N, HID);

    // 4) RoPE + head split (+ V transpose)
    rope_split_kernel<<<dim3(S_LEN, NHEADS), HDIM, 0, stream>>>(
        qkv_bf, cosb, sinb, Qb, Kb, Vtb);

    // 5) sliding-window attention -> bf16 [4096,2048]
    attn_kernel<<<dim3(S_LEN / 64, NHEADS), 128, 0, stream>>>(Qb, Kb, Vtb, attn_o);

    // 6) O projection: [4096,2048] x [2048,2048]^T -> fp32 d_out
    gemm_nt_kernel<false><<<dim3(HID / 128, S_LEN / 128), 128, 0, stream>>>(
        attn_o, ow_bf, out, S_LEN, HID, HID);
}
